// GraphormerAttention_84894323573343
// MI455X (gfx1250) — matrix operand order repaired and checked by reading, not compiled, not run on hardware
//
#include <hip/hip_runtime.h>

// Problem constants (match reference)
#define B_     4
#define N_     1024
#define DIN_   512
#define H_     8
#define DH_    64
#define INNER_ 512

typedef __attribute__((ext_vector_type(16))) __bf16 v16bf;
typedef __attribute__((ext_vector_type(8)))  float  v8f;

static __device__ __forceinline__ v8f wmma_bf16(v16bf a, v16bf b, v8f c) {
  // D = A(16x32 bf16) * B(32x16 bf16) + C(16x16 f32)
  return __builtin_amdgcn_wmma_f32_16x16x32_bf16(false, a, false, b, (short)0, c,
                                                 false, false);
}

// A-fragment: 16x32 (MxK) bf16, source row-major [row][K], ldk = row stride.
// Lane l holds row (l%16); elements 0..15 = K in [kBase+16*half, +16)
// contiguous (matches load_b_frag K order).
static __device__ __forceinline__ v16bf load_a_frag(const __bf16* base, int ldk,
                                                    int row0, int kBase, int lane) {
  const int m = row0 + (lane & 15);
  const int half = lane >> 4;
  const __bf16* p = base + (size_t)m * ldk + kBase;
  v16bf a;
#pragma unroll
  for (int i = 0; i < 8; ++i) {
    a[i]     = p[16 * half + i];
    a[i + 8] = p[16 * half + 8 + i];
  }
  return a;
}

// B-fragment: 32x16 (KxN) bf16, source stored N-major: row n holds K values,
// ldk = stride between n rows. Lane l holds col (l%16); elements 0..15 = K in
// [kBase+16*half, +16) contiguous (this code's order, same as load_a_frag).
static __device__ __forceinline__ v16bf load_b_frag(const __bf16* base, int ldk,
                                                    int col0, int kBase, int lane) {
  const int n = col0 + (lane & 15);
  const int half = lane >> 4;
  const __bf16* p = base + (size_t)n * ldk + kBase + 16 * half;
  v16bf b;
#pragma unroll
  for (int i = 0; i < 16; ++i) b[i] = p[i];
  return b;
}

// Row reductions across the 16-lane group holding one C row (wave32).
static __device__ __forceinline__ float rowred_max(float v) {
#pragma unroll
  for (int m = 1; m <= 8; m <<= 1) v = fmaxf(v, __shfl_xor(v, m, 32));
  return v;
}
static __device__ __forceinline__ float rowred_sum(float v) {
#pragma unroll
  for (int m = 1; m <= 8; m <<= 1) v += __shfl_xor(v, m, 32);
  return v;
}

// ---------------------------------------------------------------- prep kernels
__global__ void cast_x_kernel(const float* __restrict__ x, __bf16* __restrict__ xb,
                              int n) {
  int i = blockIdx.x * blockDim.x + threadIdx.x;
  if (i < n) xb[i] = (__bf16)x[i];
}

// WT[o][k] = W[k][colOff + o] (bf16), for OUT*IN = 262144 elements exactly.
__global__ void transpose_cast_kernel(const float* __restrict__ W,
                                      __bf16* __restrict__ WT,
                                      int IN, int colOff, int ld) {
  int i = blockIdx.x * blockDim.x + threadIdx.x;
  int o = i / IN, k = i - o * IN;
  WT[(size_t)o * IN + k] = (__bf16)W[(size_t)k * ld + colOff + o];
}

// --------------------------------------------------------------- QKV projection
// One wave = 32x64 output block (2 M-tiles x 4 N-tiles): 8 WMMA per K-step.
// Per which: (4096/32) * (512/64) = 128*8 = 1024 waves; x3 = 3072 waves.
__global__ void __launch_bounds__(128) qkv_proj_kernel(
    const __bf16* __restrict__ xb,
    const __bf16* __restrict__ WqT, const __bf16* __restrict__ WkT,
    const __bf16* __restrict__ WvT,
    __bf16* __restrict__ Qb, __bf16* __restrict__ Kb, __bf16* __restrict__ VTb) {
  const int lane = threadIdx.x & 31;
  const int gw = blockIdx.x * (blockDim.x >> 5) + (threadIdx.x >> 5);
  const int which = gw >> 10;          // 0=Q 1=K 2=V (wave-uniform)
  const int t = gw & 1023;
  const int m0 = (t >> 3) << 5;        // 32-row block
  const int n0 = (t & 7) << 6;         // 64-col block
  const __bf16* wt = (which == 0) ? WqT : ((which == 1) ? WkT : WvT);

  v8f c[2][4] = {};
  for (int kb = 0; kb < DIN_; kb += 32) {
    v16bf a0 = load_a_frag(xb, DIN_, m0, kb, lane);
    v16bf a1 = load_a_frag(xb, DIN_, m0 + 16, kb, lane);
    v16bf b0 = load_b_frag(wt, DIN_, n0,      kb, lane);
    v16bf b1 = load_b_frag(wt, DIN_, n0 + 16, kb, lane);
    v16bf b2 = load_b_frag(wt, DIN_, n0 + 32, kb, lane);
    v16bf b3 = load_b_frag(wt, DIN_, n0 + 48, kb, lane);
    c[0][0] = wmma_bf16(a0, b0, c[0][0]);
    c[0][1] = wmma_bf16(a0, b1, c[0][1]);
    c[0][2] = wmma_bf16(a0, b2, c[0][2]);
    c[0][3] = wmma_bf16(a0, b3, c[0][3]);
    c[1][0] = wmma_bf16(a1, b0, c[1][0]);
    c[1][1] = wmma_bf16(a1, b1, c[1][1]);
    c[1][2] = wmma_bf16(a1, b2, c[1][2]);
    c[1][3] = wmma_bf16(a1, b3, c[1][3]);
  }

  const int half = lane >> 4, nl = lane & 15;
#pragma unroll
  for (int mi = 0; mi < 2; ++mi) {
#pragma unroll
    for (int nj = 0; nj < 4; ++nj) {
#pragma unroll
      for (int r = 0; r < 8; ++r) {
        int gm = m0 + 16 * mi + r + 8 * half;  // global token row
        int gn = n0 + 16 * nj + nl;            // global feature
        int b_ = gm >> 10, tok = gm & (N_ - 1);
        int h = gn >> 6, d = gn & (DH_ - 1);
        size_t bh = (size_t)b_ * H_ + h;
        __bf16 v = (__bf16)c[mi][nj][r];
        if (which == 0)      Qb[(bh * N_ + tok) * DH_ + d] = v;
        else if (which == 1) Kb[(bh * N_ + tok) * DH_ + d] = v;
        else                 VTb[(bh * DH_ + d) * N_ + tok] = v;  // V transposed
      }
    }
  }
}

// ------------------------------------------------------- fused flash attention
// Block = 8 waves = the 8 heads of one (b, 16-query tile). All 8 waves stream
// the SAME sim_bias cachelines ([b][i][j][h], h contiguous) concurrently, so
// the 134 MB bias stream is fetched from HBM exactly once (WGP$/L2 reuse).
// Grid: B * (N/16) = 256 blocks x 256 threads.
__global__ void __launch_bounds__(256) attn_kernel(
    const __bf16* __restrict__ Qb, const __bf16* __restrict__ Kb,
    const __bf16* __restrict__ VTb, const float* __restrict__ bias,
    const unsigned char* __restrict__ mask, __bf16* __restrict__ AOb) {
  __shared__ __bf16 ldsP[8][16][40];   // per-wave 16x32 P tile, padded rows

  const int lane = threadIdx.x & 31;
  const int h = threadIdx.x >> 5;      // wave index == head
  const int b = blockIdx.x >> 6;
  const int it = blockIdx.x & 63;      // query tile
  const int bh = b * H_ + h;
  const int i0 = it << 4;
  const int half = lane >> 4, nl = lane & 15;

  const __bf16* qbase = Qb  + (size_t)bh * N_ * DH_;
  const __bf16* kbase = Kb  + (size_t)bh * N_ * DH_;
  const __bf16* vbase = VTb + (size_t)bh * DH_ * N_;

  const v16bf aq0 = load_a_frag(qbase, DH_, i0, 0, lane);
  const v16bf aq1 = load_a_frag(qbase, DH_, i0, 32, lane);

  bool mi[8];
#pragma unroll
  for (int r = 0; r < 8; ++r)
    mi[r] = mask[(size_t)b * N_ + i0 + r + 8 * half] != 0;

  float mrun[8], lrun[8];
#pragma unroll
  for (int r = 0; r < 8; ++r) { mrun[r] = -__builtin_inff(); lrun[r] = 0.f; }
  v8f acc[4] = {};

  const float scale = 0.125f;                 // DH^-0.5
  const float NEGF = -3.402823466e38f;        // finfo(float32).min

  for (int jb = 0; jb < N_; jb += 32) {
    if (jb + 32 < N_) {
      // prefetch next K block (one 128B row per lane) and next bias chunk
      __builtin_prefetch(kbase + (size_t)(jb + 32 + lane) * DH_, 0, 1);
      __builtin_prefetch(
          bias + (((size_t)b * N_ + i0 + nl) * N_ + jb + 32 + 16 * half) * H_, 0, 1);
    }

    // Two 16x16 score tiles over this 32-column block.
    v16bf bk00 = load_b_frag(kbase, DH_, jb,      0, lane);
    v16bf bk01 = load_b_frag(kbase, DH_, jb,     32, lane);
    v16bf bk10 = load_b_frag(kbase, DH_, jb + 16, 0, lane);
    v16bf bk11 = load_b_frag(kbase, DH_, jb + 16, 32, lane);
    v8f s0 = {}, s1 = {};
    s0 = wmma_bf16(aq0, bk00, s0);
    s0 = wmma_bf16(aq1, bk01, s0);
    s1 = wmma_bf16(aq0, bk10, s1);
    s1 = wmma_bf16(aq1, bk11, s1);

    const bool mj0 = mask[(size_t)b * N_ + jb + nl] != 0;
    const bool mj1 = mask[(size_t)b * N_ + jb + 16 + nl] != 0;

#pragma unroll
    for (int r = 0; r < 8; ++r) {
      const int i = i0 + r + 8 * half;
      const size_t bidx = (((size_t)b * N_ + i) * N_ + jb) * H_ + h;
      float e0 = s0[r] * scale + bias[bidx + (size_t)nl * H_];
      float e1 = s1[r] * scale + bias[bidx + (size_t)(nl + 16) * H_];
      if (!(mi[r] && mj0)) e0 = NEGF;
      if (!(mi[r] && mj1)) e1 = NEGF;

      // online softmax update for row i over these 32 columns
      float rmax = rowred_max(fmaxf(e0, e1));
      float mnew = fmaxf(mrun[r], rmax);
      float al = __expf(mrun[r] - mnew);      // 0 on first block
      float p0 = __expf(e0 - mnew);
      float p1 = __expf(e1 - mnew);
      lrun[r] = lrun[r] * al + rowred_sum(p0 + p1);
      mrun[r] = mnew;
#pragma unroll
      for (int t = 0; t < 4; ++t) acc[t][r] *= al;

      const int m = r + 8 * half;             // C-layout row -> LDS row
      ldsP[h][m][nl]      = (__bf16)p0;
      ldsP[h][m][nl + 16] = (__bf16)p1;
    }

    // P (C-layout) -> A-layout via per-wave LDS tile (DS ops in-order per wave)
    v16bf ap = load_a_frag(&ldsP[h][0][0], 40, 0, 0, lane);
#pragma unroll
    for (int t = 0; t < 4; ++t) {
      v16bf bv = load_b_frag(vbase, N_, 16 * t, jb, lane);  // V^T: contiguous j
      acc[t] = wmma_bf16(ap, bv, acc[t]);
    }
  }

  // epilogue: O /= l, store as [b][token][h*64+d] bf16 for the out projection
#pragma unroll
  for (int t = 0; t < 4; ++t) {
#pragma unroll
    for (int r = 0; r < 8; ++r) {
      int tok = i0 + r + 8 * half;
      int feat = h * DH_ + 16 * t + nl;
      AOb[((size_t)b * N_ + tok) * INNER_ + feat] = (__bf16)(acc[t][r] / lrun[r]);
    }
  }
}

// ------------------------------------------------------------ output projection
// One wave = 32x64 output block, same 2x4 register blocking. 1024 waves.
__global__ void __launch_bounds__(128) oproj_kernel(
    const __bf16* __restrict__ AOb, const __bf16* __restrict__ WoT,
    const float* __restrict__ bo, float* __restrict__ out) {
  const int lane = threadIdx.x & 31;
  const int gw = blockIdx.x * (blockDim.x >> 5) + (threadIdx.x >> 5);
  const int m0 = (gw >> 3) << 5;
  const int n0 = (gw & 7) << 6;

  v8f c[2][4] = {};
  for (int kb = 0; kb < INNER_; kb += 32) {
    v16bf a0 = load_a_frag(AOb, INNER_, m0, kb, lane);
    v16bf a1 = load_a_frag(AOb, INNER_, m0 + 16, kb, lane);
    v16bf b0 = load_b_frag(WoT, INNER_, n0,      kb, lane);
    v16bf b1 = load_b_frag(WoT, INNER_, n0 + 16, kb, lane);
    v16bf b2 = load_b_frag(WoT, INNER_, n0 + 32, kb, lane);
    v16bf b3 = load_b_frag(WoT, INNER_, n0 + 48, kb, lane);
    c[0][0] = wmma_bf16(a0, b0, c[0][0]);
    c[0][1] = wmma_bf16(a0, b1, c[0][1]);
    c[0][2] = wmma_bf16(a0, b2, c[0][2]);
    c[0][3] = wmma_bf16(a0, b3, c[0][3]);
    c[1][0] = wmma_bf16(a1, b0, c[1][0]);
    c[1][1] = wmma_bf16(a1, b1, c[1][1]);
    c[1][2] = wmma_bf16(a1, b2, c[1][2]);
    c[1][3] = wmma_bf16(a1, b3, c[1][3]);
  }

  const int half = lane >> 4, nl = lane & 15;
#pragma unroll
  for (int mi = 0; mi < 2; ++mi) {
#pragma unroll
    for (int nj = 0; nj < 4; ++nj) {
      int gn = n0 + 16 * nj + nl;
      float bv = bo[gn];
#pragma unroll
      for (int r = 0; r < 8; ++r) {
        int gm = m0 + 16 * mi + r + 8 * half;
        out[(size_t)gm * DIN_ + gn] = c[mi][nj][r] + bv;
      }
    }
  }
}

// --------------------------------------------------------------------- launcher
extern "C" void kernel_launch(void* const* d_in, const int* in_sizes, int n_in,
                              void* d_out, int out_size, void* d_ws, size_t ws_size,
                              hipStream_t stream) {
  const float*         x    = (const float*)d_in[0];
  const unsigned char* mask = (const unsigned char*)d_in[1]; // jnp.bool_ (1 byte)
  const float*         bias = (const float*)d_in[2];
  const float*         Wq   = (const float*)d_in[3];
  const float*         Wkv  = (const float*)d_in[4];
  const float*         Wo   = (const float*)d_in[5];
  const float*         bo   = (const float*)d_in[6];
  float*               out  = (float*)d_out;

  // Workspace layout (bf16 elements). Total ~22 MB.
  __bf16* ws  = (__bf16*)d_ws;
  __bf16* xb  = ws;                     // [4096][512]
  __bf16* Qb  = xb  + 2097152;          // [B][H][N][DH]
  __bf16* Kb  = Qb  + 2097152;          // [B][H][N][DH]
  __bf16* VTb = Kb  + 2097152;          // [B][H][DH][N]
  __bf16* AOb = VTb + 2097152;          // [4096][512]
  __bf16* WqT = AOb + 2097152;          // [512][512]
  __bf16* WkT = WqT + 262144;
  __bf16* WvT = WkT + 262144;
  __bf16* WoT = WvT + 262144;

  cast_x_kernel<<<8192, 256, 0, stream>>>(x, xb, 2097152);
  transpose_cast_kernel<<<1024, 256, 0, stream>>>(Wq,  WqT, 512, 0,   512);
  transpose_cast_kernel<<<1024, 256, 0, stream>>>(Wkv, WkT, 512, 0,   1024);
  transpose_cast_kernel<<<1024, 256, 0, stream>>>(Wkv, WvT, 512, 512, 1024);
  transpose_cast_kernel<<<1024, 256, 0, stream>>>(Wo,  WoT, 512, 0,   512);

  // 3072 waves / 4 per block
  qkv_proj_kernel<<<768, 128, 0, stream>>>(xb, WqT, WkT, WvT, Qb, Kb, VTb);
  // 8 heads per block, B*(N/16) blocks
  attn_kernel<<<256, 256, 0, stream>>>(Qb, Kb, VTb, bias, mask, AOb);
  // 1024 waves / 4 per block
  oproj_kernel<<<256, 128, 0, stream>>>(AOb, WoT, bo, out);
}
